// PipelineGPT2_13898514170532
// MI455X (gfx1250) — matrix operand order repaired and checked by reading, not compiled, not run on hardware
//
#include <hip/hip_runtime.h>
#include <hip/hip_bf16.h>

// ---------------------------------------------------------------------------
// GPT-2 forward on gfx1250 (MI455X): bf16 WMMA GEMMs + flash attention.
// All matrix math via V_WMMA_F32_16X16X32_BF16 (fp32 accumulate).
// GEMM: 64x128 block tile, 8 waves, wave = 16x64 output (1 A-frag -> 4 WMMA
// per K-step), double-buffered LDS (1 barrier / K-step), global loads for
// tile k+1 overlapped with WMMAs of tile k.  fp32->bf16 packing via the
// hardware v_cvt_pk_bf16_f32 instruction (inline asm; clang-22 lacks the
// builtin for gfx1250).
// ---------------------------------------------------------------------------

typedef __bf16 bf16_t;
typedef bf16_t v16bf __attribute__((ext_vector_type(16)));
typedef float  v8f   __attribute__((ext_vector_type(8)));

union Frag16 {
    v16bf v;
    uint4 q[2];
    unsigned short u[16];
};

__device__ __forceinline__ unsigned short f2bf(float f) {
    unsigned u = __builtin_bit_cast(unsigned, f);
    return (unsigned short)((u + 0x7FFFu + ((u >> 16) & 1u)) >> 16);  // RNE
}

// Packed fp32 -> 2 x bf16 (round-to-nearest-even) in one VALU op.
__device__ __forceinline__ unsigned pk2(float a, float b) {
    unsigned r;
    asm("v_cvt_pk_bf16_f32 %0, %1, %2" : "=v"(r) : "v"(a), "v"(b));
    return r;
}

__device__ __forceinline__ uint4 pack8(float4 a, float4 b) {
    uint4 r;
    r.x = pk2(a.x, a.y); r.y = pk2(a.z, a.w);
    r.z = pk2(b.x, b.y); r.w = pk2(b.z, b.w);
    return r;
}
__device__ __forceinline__ float gelu_f(float x) {
    float u = 0.7978845608028654f * (x + 0.044715f * x * x * x);
    return x / (1.0f + __expf(-2.0f * u));   // 0.5x(1+tanh(u))
}

// ------------------------------ embeddings ---------------------------------
__global__ __launch_bounds__(256) void embed_kernel(
    const int* __restrict__ idx, const float* __restrict__ wte,
    const float* __restrict__ wpe, float* __restrict__ x)
{
    int row = blockIdx.x;            // b*T + t
    int t   = row & 1023;
    int tok = idx[row];
    for (int c = threadIdx.x; c < 768; c += 256)
        x[(size_t)row * 768 + c] = wte[(size_t)tok * 768 + c] + wpe[(size_t)t * 768 + c];
}

// ------------------------------ layernorm ----------------------------------
__global__ __launch_bounds__(256) void ln_kernel(
    const float* __restrict__ x, const float* __restrict__ w,
    const float* __restrict__ bb, float* __restrict__ y)
{
    __shared__ float r1[256], r2[256];
    int row = blockIdx.x;
    const float* xr = x + (size_t)row * 768;
    int i0 = threadIdx.x, i1 = i0 + 256, i2 = i0 + 512;
    float v0 = xr[i0], v1 = xr[i1], v2 = xr[i2];
    r1[i0] = v0 + v1 + v2;
    r2[i0] = v0 * v0 + v1 * v1 + v2 * v2;
    __syncthreads();
    for (int off = 128; off > 0; off >>= 1) {
        if (threadIdx.x < off) {
            r1[threadIdx.x] += r1[threadIdx.x + off];
            r2[threadIdx.x] += r2[threadIdx.x + off];
        }
        __syncthreads();
    }
    float mu  = r1[0] * (1.0f / 768.0f);
    float var = r2[0] * (1.0f / 768.0f) - mu * mu;
    float rs  = rsqrtf(var + 1e-5f);
    float* yr = y + (size_t)row * 768;
    yr[i0] = (v0 - mu) * rs * w[i0] + bb[i0];
    yr[i1] = (v1 - mu) * rs * w[i1] + bb[i1];
    yr[i2] = (v2 - mu) * rs * w[i2] + bb[i2];
}

// ------------------------------ WMMA GEMM ----------------------------------
// Y[M,N] = act(X[M,K] @ W[N,K]^T + bias) (+ res).
#define APITCH 48   // 32 bf16 + pad, keeps ds_load_b128 16B-aligned

__global__ __launch_bounds__(256) void gemm_kernel(
    const float* __restrict__ X, const float* __restrict__ W,
    const float* __restrict__ bias, const float* __restrict__ res,
    float* __restrict__ Y, int M, int N, int K, int act)
{
    __shared__ unsigned short a_lds[2][64  * APITCH];
    __shared__ unsigned short b_lds[2][128 * APITCH];
    const int tid  = threadIdx.x;
    const int lane = tid & 31, wid = tid >> 5;
    const int half = lane >> 4, l16 = lane & 15;
    const int m0 = blockIdx.y * 64, n0 = blockIdx.x * 128;
    const int ar = tid >> 2;            // A loader row 0..63
    const int ac = (tid & 3) * 8;       // A loader col 0,8,16,24
    const int br = tid >> 1;            // B loader row 0..127
    const int bc = (tid & 1) * 16;      // B loader col 0,16
    const int mrow = (wid >> 1) * 16;   // wave M sub-tile (4)
    const int ncol = (wid & 1) * 64;    // wave N sub-tile (2), 4x16 inside

    const float* aptr = X + (size_t)(m0 + ar) * K + ac;
    const float* bptr = W + (size_t)(n0 + br) * K + bc;
    const int a_off0 = (mrow + l16) * APITCH + half * 16;
    const int a_off1 = a_off0 + 8;
    const int b_off  = (ncol + l16) * APITCH + half * 16;

    // ---- prologue: stage K-tile 0 into buffer 0
    {
        float4 a0 = *(const float4*)aptr,       a1 = *(const float4*)(aptr + 4);
        float4 b0 = *(const float4*)bptr,       b1 = *(const float4*)(bptr + 4);
        float4 b2 = *(const float4*)(bptr + 8), b3 = *(const float4*)(bptr + 12);
        *(uint4*)&a_lds[0][ar * APITCH + ac]     = pack8(a0, a1);
        *(uint4*)&b_lds[0][br * APITCH + bc]     = pack8(b0, b1);
        *(uint4*)&b_lds[0][br * APITCH + bc + 8] = pack8(b2, b3);
    }

    v8f c0 = {}, c1 = {}, c2 = {}, c3 = {};
    int cur = 0;
    // ---- steady state: compute tile k while fetching tile k+1
    for (int k0 = 0; k0 + 32 < K; k0 += 32, cur ^= 1) {
        __syncthreads();
        const float* ap = aptr + k0 + 32;
        const float* bp = bptr + k0 + 32;
        float4 a0 = *(const float4*)ap,       a1 = *(const float4*)(ap + 4);
        float4 b0 = *(const float4*)bp,       b1 = *(const float4*)(bp + 4);
        float4 b2 = *(const float4*)(bp + 8), b3 = *(const float4*)(bp + 12);
        __builtin_prefetch(ap + 32, 0, 3);   // global_prefetch_b8 (speculative)
        __builtin_prefetch(bp + 32, 0, 3);

        Frag16 a, f0, f1, f2, f3;
        a.q[0]  = *(const uint4*)&a_lds[cur][a_off0];
        a.q[1]  = *(const uint4*)&a_lds[cur][a_off1];
        f0.q[0] = *(const uint4*)&b_lds[cur][b_off];
        f0.q[1] = *(const uint4*)&b_lds[cur][b_off + 8];
        f1.q[0] = *(const uint4*)&b_lds[cur][b_off + 16 * APITCH];
        f1.q[1] = *(const uint4*)&b_lds[cur][b_off + 16 * APITCH + 8];
        f2.q[0] = *(const uint4*)&b_lds[cur][b_off + 32 * APITCH];
        f2.q[1] = *(const uint4*)&b_lds[cur][b_off + 32 * APITCH + 8];
        f3.q[0] = *(const uint4*)&b_lds[cur][b_off + 48 * APITCH];
        f3.q[1] = *(const uint4*)&b_lds[cur][b_off + 48 * APITCH + 8];

        c0 = __builtin_amdgcn_wmma_f32_16x16x32_bf16(false, a.v, false, f0.v, (short)0, c0, false, false);
        c1 = __builtin_amdgcn_wmma_f32_16x16x32_bf16(false, a.v, false, f1.v, (short)0, c1, false, false);
        c2 = __builtin_amdgcn_wmma_f32_16x16x32_bf16(false, a.v, false, f2.v, (short)0, c2, false, false);
        c3 = __builtin_amdgcn_wmma_f32_16x16x32_bf16(false, a.v, false, f3.v, (short)0, c3, false, false);

        *(uint4*)&a_lds[cur ^ 1][ar * APITCH + ac]     = pack8(a0, a1);
        *(uint4*)&b_lds[cur ^ 1][br * APITCH + bc]     = pack8(b0, b1);
        *(uint4*)&b_lds[cur ^ 1][br * APITCH + bc + 8] = pack8(b2, b3);
    }
    // ---- final tile
    __syncthreads();
    {
        Frag16 a, f0, f1, f2, f3;
        a.q[0]  = *(const uint4*)&a_lds[cur][a_off0];
        a.q[1]  = *(const uint4*)&a_lds[cur][a_off1];
        f0.q[0] = *(const uint4*)&b_lds[cur][b_off];
        f0.q[1] = *(const uint4*)&b_lds[cur][b_off + 8];
        f1.q[0] = *(const uint4*)&b_lds[cur][b_off + 16 * APITCH];
        f1.q[1] = *(const uint4*)&b_lds[cur][b_off + 16 * APITCH + 8];
        f2.q[0] = *(const uint4*)&b_lds[cur][b_off + 32 * APITCH];
        f2.q[1] = *(const uint4*)&b_lds[cur][b_off + 32 * APITCH + 8];
        f3.q[0] = *(const uint4*)&b_lds[cur][b_off + 48 * APITCH];
        f3.q[1] = *(const uint4*)&b_lds[cur][b_off + 48 * APITCH + 8];
        c0 = __builtin_amdgcn_wmma_f32_16x16x32_bf16(false, a.v, false, f0.v, (short)0, c0, false, false);
        c1 = __builtin_amdgcn_wmma_f32_16x16x32_bf16(false, a.v, false, f1.v, (short)0, c1, false, false);
        c2 = __builtin_amdgcn_wmma_f32_16x16x32_bf16(false, a.v, false, f2.v, (short)0, c2, false, false);
        c3 = __builtin_amdgcn_wmma_f32_16x16x32_bf16(false, a.v, false, f3.v, (short)0, c3, false, false);
    }

    // ---- epilogue (uniform branches hoisted out of element loops)
    v8f cc[4] = {c0, c1, c2, c3};
#pragma unroll
    for (int j = 0; j < 4; ++j) {
        int gc = n0 + ncol + j * 16 + l16;
        float bia = bias ? bias[gc] : 0.0f;
        float vv[8];
#pragma unroll
        for (int r = 0; r < 8; ++r) vv[r] = cc[j][r] + bia;
        if (act) {
#pragma unroll
            for (int r = 0; r < 8; ++r) vv[r] = gelu_f(vv[r]);
        }
        if (res) {
#pragma unroll
            for (int r = 0; r < 8; ++r)
                vv[r] += res[(size_t)(m0 + mrow + r + half * 8) * N + gc];
        }
#pragma unroll
        for (int r = 0; r < 8; ++r)
            Y[(size_t)(m0 + mrow + r + half * 8) * N + gc] = vv[r];
    }
}

// --------------------------- flash attention -------------------------------
// One wave per 16-row Q tile of one (b,h).  S = (Q/8)K^T and O += P V via
// bf16 WMMA, online softmax in f32.  P staged via per-wave LDS to transpose
// C-layout -> A-fragment layout (in-order DS within wave, no barrier needed).
__global__ __launch_bounds__(256) void attn_kernel(
    const float* __restrict__ qkv, float* __restrict__ atty)
{
    __shared__ unsigned short p_lds[8][16 * 32];
    const int lane = threadIdx.x & 31, wid = threadIdx.x >> 5;
    const int half = lane >> 4, l16 = lane & 15;
    const int gw = blockIdx.x * 8 + wid;
    const int qt = gw & 63;          // T/16 = 64 q-tiles
    const int bh = gw >> 6;
    const int b  = bh / 12, h = bh % 12;
    const int q0 = qt * 16;
    const size_t ldq = 2304;
    const float* qptr = qkv + (size_t)b * 1024 * ldq + h * 64;
    const float* kptr = qptr + 768;
    const float* vptr = qptr + 1536;

    // Q fragments for this wave's 16 rows, pre-scaled by 1/sqrt(D)=0.125
    Frag16 aq[2];
    {
        const float* qr = qptr + (size_t)(q0 + l16) * ldq;
        const float s = 0.125f;
        for (int ds = 0; ds < 2; ++ds) {
            const float* p0 = qr + ds * 32 + half * 16;
            const float* p1 = qr + ds * 32 + half * 16 + 8;
            float4 x0 = *(const float4*)p0, x1 = *(const float4*)(p0 + 4);
            float4 y0 = *(const float4*)p1, y1 = *(const float4*)(p1 + 4);
            x0.x*=s; x0.y*=s; x0.z*=s; x0.w*=s;  x1.x*=s; x1.y*=s; x1.z*=s; x1.w*=s;
            y0.x*=s; y0.y*=s; y0.z*=s; y0.w*=s;  y1.x*=s; y1.y*=s; y1.z*=s; y1.w*=s;
            aq[ds].q[0] = pack8(x0, x1);
            aq[ds].q[1] = pack8(y0, y1);
        }
    }

    float m8[8], l8[8];
    v8f o[4] = {};
    for (int r = 0; r < 8; ++r) { m8[r] = -3.0e38f; l8[r] = 0.0f; }

    const int nch = (q0 + 16 + 31) >> 5;        // causal 32-col chunks
    for (int ch = 0; ch < nch; ++ch) {
        const int kb = ch * 32;
        v8f s0 = {}, s1 = {};
#pragma unroll
        for (int ds = 0; ds < 2; ++ds) {
            Frag16 bk0, bk1;
            const float* kr0 = kptr + (size_t)(kb + l16) * ldq + ds * 32 + half * 16;
            const float* kr1 = kptr + (size_t)(kb + 16 + l16) * ldq + ds * 32 + half * 16;
            bk0.q[0] = pack8(*(const float4*)kr0, *(const float4*)(kr0 + 4));
            bk0.q[1] = pack8(*(const float4*)(kr0 + 8), *(const float4*)(kr0 + 12));
            bk1.q[0] = pack8(*(const float4*)kr1, *(const float4*)(kr1 + 4));
            bk1.q[1] = pack8(*(const float4*)(kr1 + 8), *(const float4*)(kr1 + 12));
            s0 = __builtin_amdgcn_wmma_f32_16x16x32_bf16(false, aq[ds].v, false, bk0.v,
                                                         (short)0, s0, false, false);
            s1 = __builtin_amdgcn_wmma_f32_16x16x32_bf16(false, aq[ds].v, false, bk1.v,
                                                         (short)0, s1, false, false);
        }
        // causal mask + online softmax (per C-layout row r+8*half)
        const int col0 = kb + l16, col1 = kb + 16 + l16;
        for (int r = 0; r < 8; ++r) {
            int row = q0 + r + half * 8;
            float t0 = (col0 <= row) ? s0[r] : -3.0e38f;
            float t1 = (col1 <= row) ? s1[r] : -3.0e38f;
            float mx = fmaxf(t0, t1);
            for (int sh = 8; sh >= 1; sh >>= 1) mx = fmaxf(mx, __shfl_xor(mx, sh, 32));
            float mn = fmaxf(m8[r], mx);
            float sc = __expf(m8[r] - mn);
            float e0 = __expf(t0 - mn), e1 = __expf(t1 - mn);
            float sum = e0 + e1;
            for (int sh = 8; sh >= 1; sh >>= 1) sum += __shfl_xor(sum, sh, 32);
            l8[r] = l8[r] * sc + sum;
            m8[r] = mn;
            o[0][r] *= sc; o[1][r] *= sc; o[2][r] *= sc; o[3][r] *= sc;
            p_lds[wid][(r + half * 8) * 32 + l16]      = f2bf(e0);
            p_lds[wid][(r + half * 8) * 32 + 16 + l16] = f2bf(e1);
        }
        // P as A-fragment (16x32)
        Frag16 pf;
        pf.q[0] = *(const uint4*)&p_lds[wid][l16 * 32 + half * 16];
        pf.q[1] = *(const uint4*)&p_lds[wid][l16 * 32 + half * 16 + 8];
        // O += P @ V  (4 d-subtiles of 16)
#pragma unroll
        for (int j = 0; j < 4; ++j) {
            Frag16 bv;
            for (int i = 0; i < 16; ++i) {
                float vv = vptr[(size_t)(kb + half * 16 + i) * ldq + j * 16 + l16];
                bv.u[i] = f2bf(vv);
            }
            o[j] = __builtin_amdgcn_wmma_f32_16x16x32_bf16(false, pf.v, false, bv.v,
                                                           (short)0, o[j], false, false);
        }
    }

    for (int j = 0; j < 4; ++j)
        for (int r = 0; r < 8; ++r) {
            int row = q0 + r + half * 8;
            atty[((size_t)b * 1024 + row) * 768 + h * 64 + j * 16 + l16] = o[j][r] / l8[r];
        }
}

// ------------------------------- driver ------------------------------------
extern "C" void kernel_launch(void* const* d_in, const int* in_sizes, int n_in,
                              void* d_out, int out_size, void* d_ws, size_t ws_size,
                              hipStream_t stream) {
    const int*   idx     = (const int*)  d_in[0];
    const float* wte     = (const float*)d_in[1];
    const float* wpe     = (const float*)d_in[2];
    const float* ln1w    = (const float*)d_in[3];
    const float* ln1b    = (const float*)d_in[4];
    const float* qkvw    = (const float*)d_in[5];
    const float* qkvb    = (const float*)d_in[6];
    const float* projw   = (const float*)d_in[7];
    const float* projb   = (const float*)d_in[8];
    const float* ln2w    = (const float*)d_in[9];
    const float* ln2b    = (const float*)d_in[10];
    const float* fcw     = (const float*)d_in[11];
    const float* fcb     = (const float*)d_in[12];
    const float* fcprojw = (const float*)d_in[13];
    const float* fcprojb = (const float*)d_in[14];
    const float* lnfw    = (const float*)d_in[15];
    const float* lnfb    = (const float*)d_in[16];
    float* out = (float*)d_out;

    const int BT = 2048, C = 768;
    float* ws   = (float*)d_ws;
    float* x    = ws;                         // 2048*768
    float* ln   = x    + (size_t)BT * C;      // 2048*768
    float* qkv  = ln   + (size_t)BT * C;      // 2048*2304
    float* atty = qkv  + (size_t)BT * 3 * C;  // 2048*768
    float* h2   = atty + (size_t)BT * C;      // 2048*3072

    embed_kernel<<<BT, 256, 0, stream>>>(idx, wte, wpe, x);

    for (int l = 0; l < 12; ++l) {
        ln_kernel<<<BT, 256, 0, stream>>>(x, ln1w + l * C, ln1b + l * C, ln);
        gemm_kernel<<<dim3(2304 / 128, BT / 64), 256, 0, stream>>>(
            ln, qkvw + (size_t)l * 3 * C * C, qkvb + l * 3 * C, nullptr,
            qkv, BT, 2304, C, 0);
        attn_kernel<<<192, 256, 0, stream>>>(qkv, atty);
        gemm_kernel<<<dim3(C / 128, BT / 64), 256, 0, stream>>>(
            atty, projw + (size_t)l * C * C, projb + l * C, x,
            x, BT, C, C, 0);
        ln_kernel<<<BT, 256, 0, stream>>>(x, ln2w + l * C, ln2b + l * C, ln);
        gemm_kernel<<<dim3(3072 / 128, BT / 64), 256, 0, stream>>>(
            ln, fcw + (size_t)l * 4 * C * C, fcb + l * 4 * C, nullptr,
            h2, BT, 3072, C, 1);
        gemm_kernel<<<dim3(C / 128, BT / 64), 256, 0, stream>>>(
            h2, fcprojw + (size_t)l * C * 4 * C, fcprojb + l * C, x,
            x, BT, C, 4 * C, 0);
    }

    ln_kernel<<<BT, 256, 0, stream>>>(x, lnfw, lnfb, ln);
    gemm_kernel<<<dim3(50304 / 128, BT / 64), 256, 0, stream>>>(
        ln, wte, nullptr, nullptr, out, BT, 50304, C, 0);
}
